// Attention_81544249082540
// MI455X (gfx1250) — matrix operand order repaired and checked by reading, not compiled, not run on hardware
//
#include <hip/hip_runtime.h>

typedef _Float16 half_t;
typedef __attribute__((ext_vector_type(16))) _Float16 v16h;
typedef __attribute__((ext_vector_type(8)))  _Float16 v8h;
typedef __attribute__((ext_vector_type(8)))  float    v8f;

#define HH   12
#define BB   4
#define SEQL 2048
#define DM   768
#define DK   64
#define NROW (BB*SEQL)     // 8192
#define NSTAGE (DM/64)     // 12 staged 64-wide k-chunks

// scheduler hint: batch DS reads ahead of the WMMA group
#if defined(__has_builtin)
#if __has_builtin(__builtin_amdgcn_sched_group_barrier)
#define SCHED_DS_THEN_WMMA()                                   \
  do {                                                         \
    __builtin_amdgcn_sched_group_barrier(0x100, 8, 0); /*DSr*/ \
    __builtin_amdgcn_sched_group_barrier(0x008, 4, 0); /*WMMA*/\
  } while (0)
#endif
#endif
#ifndef SCHED_DS_THEN_WMMA
#define SCHED_DS_THEN_WMMA() do {} while (0)
#endif

// ---------- CDNA5 async copy global->LDS (ASYNCcnt path) ----------
__device__ __forceinline__ void async_ld_b128(void* lds_ptr, const void* gptr) {
  unsigned lds_off = (unsigned)(unsigned long long)lds_ptr;   // addr[31:0] = LDS offset
  unsigned long long ga = (unsigned long long)gptr;
  asm volatile("global_load_async_to_lds_b128 %0, %1, off"
               :: "v"(lds_off), "v"(ga) : "memory");
}
__device__ __forceinline__ void wait_async0() {
  asm volatile("s_wait_asynccnt 0" ::: "memory");
}

// ---------- WMMA fragment helpers (CDNA5 16x16x32 f16, wave32) ----------
// A (16x32 f16): lane m = l&15, sel = l>>4; two contiguous 8-half chunks.
__device__ __forceinline__ v16h frag_a_f16(const half_t* base, int ld, int k0, int lane) {
  int m   = lane & 15;
  int sel = (lane >> 4) & 1;
  const half_t* p = base + (long long)m * ld + k0 + sel * 8;
  v8h lo = *(const v8h*)(p);
  v8h hi = *(const v8h*)(p + 16);
  v16h a;
#pragma unroll
  for (int i = 0; i < 8; ++i) { a[i] = lo[i]; a[i + 8] = hi[i]; }
  return a;
}

__device__ __forceinline__ v16h frag_a_f32(const float* base, int ld, int k0, int lane) {
  int m   = lane & 15;
  int sel = (lane >> 4) & 1;
  const float* p = base + (long long)m * ld + k0 + sel * 8;
  v16h a;
#pragma unroll
  for (int i = 0; i < 8; ++i) { a[i] = (half_t)p[i]; a[i + 8] = (half_t)p[16 + i]; }
  return a;
}

// B (32x16 f16) supplied as B^T row-major: bt[n][k] (row stride ld).
__device__ __forceinline__ v16h frag_b_t(const half_t* bt, int ld, int k0, int lane) {
  int n  = lane & 15;
  int kh = ((lane >> 4) & 1) * 8;
  const half_t* p = bt + (long long)n * ld + k0 + kh;
  v8h lo = *(const v8h*)(p);
  v8h hi = *(const v8h*)(p + 16);
  v16h b;
#pragma unroll
  for (int i = 0; i < 8; ++i) { b[i] = lo[i]; b[i + 8] = hi[i]; }
  return b;
}

// B fragment from an LDS-staged 64x64 chunk: rows n0..n0+15, ld = 64 halves.
__device__ __forceinline__ v16h frag_b_lds(const half_t* bt, int k0, int lane) {
  int n  = lane & 15;
  int kh = ((lane >> 4) & 1) * 8;
  const half_t* p = bt + n * 64 + k0 + kh;
  v8h lo = *(const v8h*)(p);
  v8h hi = *(const v8h*)(p + 16);
  v16h b;
#pragma unroll
  for (int i = 0; i < 8; ++i) { b[i] = lo[i]; b[i + 8] = hi[i]; }
  return b;
}

__device__ __forceinline__ v8f wmma16(v16h a, v16h b, v8f c) {
  return __builtin_amdgcn_wmma_f32_16x16x32_f16(false, a, false, b, (short)0, c, false, false);
}

// ---------- Kernel 1: batched fp32 -> f16 transpose (weights) ----------
__global__ __launch_bounds__(256) void k_transpose(const float* __restrict__ src,
                                                   half_t* __restrict__ dst,
                                                   int rows, int cols, long long total) {
  long long i = (long long)blockIdx.x * blockDim.x + threadIdx.x;
  if (i >= total) return;
  int c = (int)(i % cols);
  long long t = i / cols;
  int r = (int)(t % rows);
  long long m = t / rows;
  dst[(m * cols + c) * rows + r] = (half_t)src[i];
}

// ---------- Kernel 2: QKV projection, weights double-buffered via async->LDS ----------
__global__ __launch_bounds__(256) void k_qkv(const float* __restrict__ resid,
                                             const half_t* __restrict__ wqT,
                                             const half_t* __restrict__ wkT,
                                             const half_t* __restrict__ wvT,
                                             half_t* __restrict__ Qb,
                                             half_t* __restrict__ Kb,
                                             half_t* __restrict__ VTb) {
  __shared__ half_t wlds[2][DK][64];           // 16 KB double buffer (64x64-half stage)
  int tid  = threadIdx.x;
  int lane = tid & 31;
  int wv   = tid >> 5;
  int gw = blockIdx.x * 8 + wv;
  int rt   = gw % (NROW / 16);
  int t2   = gw / (NROW / 16);
  int h    = t2 % HH;
  int proj = t2 / HH;              // 0=Q 1=K 2=V (uniform per block)

  const half_t* wT = ((proj == 0) ? wqT : (proj == 1) ? wkT : wvT) + (long long)h * DK * DM;
  const float* arow = resid + (long long)rt * 16 * DM;

  // copy role: 256 threads x 32B = one 64x64-half chunk (8 KB) per stage
  int cn = tid >> 2;               // weight row 0..63
  int cp = (tid & 3) * 16;         // 16-half sub-chunk

  v8f c[4];
#pragma unroll
  for (int ct = 0; ct < 4; ++ct)
#pragma unroll
    for (int r = 0; r < 8; ++r) c[ct][r] = 0.f;

  // prologue: stage chunk 0 into buffer 0
  async_ld_b128(&wlds[0][cn][cp],     wT + (long long)cn * DM + cp);
  async_ld_b128(&wlds[0][cn][cp + 8], wT + (long long)cn * DM + cp + 8);

  for (int s = 0; s < NSTAGE; ++s) {
    int buf = s & 1;
    wait_async0();
    __syncthreads();
    if (s + 1 < NSTAGE) {
      const half_t* g = wT + (long long)cn * DM + (s + 1) * 64 + cp;
      async_ld_b128(&wlds[buf ^ 1][cn][cp],     g);
      async_ld_b128(&wlds[buf ^ 1][cn][cp + 8], g + 8);
    }
#pragma unroll
    for (int kk = 0; kk < 2; ++kk) {
      v16h a = frag_a_f32(arow, DM, s * 64 + kk * 32, lane);
      v16h b0 = frag_b_lds(&wlds[buf][0][0],  kk * 32, lane);
      v16h b1 = frag_b_lds(&wlds[buf][16][0], kk * 32, lane);
      v16h b2 = frag_b_lds(&wlds[buf][32][0], kk * 32, lane);
      v16h b3 = frag_b_lds(&wlds[buf][48][0], kk * 32, lane);
      c[0] = wmma16(a, b0, c[0]);
      c[1] = wmma16(a, b1, c[1]);
      c[2] = wmma16(a, b2, c[2]);
      c[3] = wmma16(a, b3, c[3]);
      SCHED_DS_THEN_WMMA();
    }
  }

  int n = lane & 15, sel = (lane >> 4) & 1;
  float scale = (proj == 0) ? 0.125f : 1.0f;   // 1/sqrt(DK) folded into Q
#pragma unroll
  for (int ct = 0; ct < 4; ++ct) {
#pragma unroll
    for (int r = 0; r < 8; ++r) {
      int row = rt * 16 + r + sel * 8;         // flattened (b,s)
      int col = ct * 16 + n;
      half_t v = (half_t)(c[ct][r] * scale);
      if (proj == 2) {
        int bb = row >> 11, ss = row & (SEQL - 1);
        VTb[(((long long)h * BB + bb) * DK + col) * SEQL + ss] = v;
      } else if (proj == 0) {
        Qb[((long long)h * NROW + row) * DK + col] = v;
      } else {
        Kb[((long long)h * NROW + row) * DK + col] = v;
      }
    }
  }
}

// ---------- Kernel 3: causal flash attention (one wave = 16 queries) ----------
__global__ __launch_bounds__(256) void k_flash(const half_t* __restrict__ Qb,
                                               const half_t* __restrict__ Kb,
                                               const half_t* __restrict__ VTb,
                                               half_t* __restrict__ multi) {
  __shared__ half_t ptile[8][16][32];          // per-wave P staging (8 KB)
  int lane = threadIdx.x & 31;
  int wv = threadIdx.x >> 5;
  int gw = blockIdx.x * 8 + wv;
  int qt = gw % (SEQL / 16);
  int t2 = gw / (SEQL / 16);
  int b  = t2 % BB;
  int h  = t2 / BB;
  int q0 = qt * 16;

  const half_t* Qh = Qb + ((long long)h * NROW + (long long)b * SEQL + q0) * DK;
  const half_t* Kh = Kb + ((long long)h * NROW + (long long)b * SEQL) * DK;
  const half_t* Vt = VTb + ((long long)h * BB + b) * (long long)DK * SEQL;

  v16h qa0 = frag_a_f16(Qh, DK, 0, lane);      // Q already pre-scaled
  v16h qa1 = frag_a_f16(Qh, DK, 32, lane);

  v8f o[4];
  float mrow[8], lrow[8];
#pragma unroll
  for (int ct = 0; ct < 4; ++ct)
#pragma unroll
    for (int r = 0; r < 8; ++r) o[ct][r] = 0.f;
#pragma unroll
  for (int r = 0; r < 8; ++r) { mrow[r] = -1e30f; lrow[r] = 0.f; }

  int n = lane & 15, sel = (lane >> 4) & 1;
  int nblk = (q0 + 15) / 32 + 1;               // key blocks of 32, causal

  for (int kb = 0; kb < nblk; ++kb) {
    int kbase = kb * 32;
    bool have1 = (kbase + 16) <= (q0 + 15);    // uniform per wave

    // prefetch next key block (speculative; harmless at the tail)
    if (kb + 1 < nblk) {
      __builtin_prefetch(Kh + (long long)(kbase + 32 + lane) * DK, 0, 1);
      __builtin_prefetch(Vt + (long long)lane * SEQL + kbase + 32, 0, 1);
      __builtin_prefetch(Vt + (long long)(32 + lane) * SEQL + kbase + 32, 0, 1);
    }

    // issue all K-tile fragment loads up front, then the score WMMAs
    v8f s0, s1;
    {
      v16h kb0 = frag_b_t(Kh + (long long)kbase * DK, DK, 0, lane);
      v16h kb1 = frag_b_t(Kh + (long long)kbase * DK, DK, 32, lane);
      v16h kc0, kc1;
      if (have1) {
        kc0 = frag_b_t(Kh + (long long)(kbase + 16) * DK, DK, 0, lane);
        kc1 = frag_b_t(Kh + (long long)(kbase + 16) * DK, DK, 32, lane);
      }
      v8f z;
#pragma unroll
      for (int r = 0; r < 8; ++r) z[r] = 0.f;
      s0 = wmma16(qa0, kb0, z);
      s0 = wmma16(qa1, kb1, s0);
      if (have1) {
        s1 = wmma16(qa0, kc0, z);
        s1 = wmma16(qa1, kc1, s1);
      } else {
#pragma unroll
        for (int r = 0; r < 8; ++r) s1[r] = -1e30f;
      }
    }

    // causal mask on diagonal tiles
    if (kbase + 15 > q0) {
#pragma unroll
      for (int r = 0; r < 8; ++r) {
        int q = q0 + r + sel * 8;
        if (kbase + n > q) s0[r] = -1e30f;
      }
    }
    if (have1 && (kbase + 16 + 15 > q0)) {
#pragma unroll
      for (int r = 0; r < 8; ++r) {
        int q = q0 + r + sel * 8;
        if (kbase + 16 + n > q) s1[r] = -1e30f;
      }
    }

    // online softmax: row max / sums across 16-lane N groups
    float p0v[8], p1v[8];
#pragma unroll
    for (int r = 0; r < 8; ++r) {
      float mx = fmaxf(s0[r], s1[r]);
#pragma unroll
      for (int d = 1; d < 16; d <<= 1) mx = fmaxf(mx, __shfl_xor(mx, d, 32));
      float nm = fmaxf(mrow[r], mx);
      float al = __expf(mrow[r] - nm);
      float p0 = __expf(s0[r] - nm);
      float p1 = have1 ? __expf(s1[r] - nm) : 0.f;
      float sm = p0 + p1;
#pragma unroll
      for (int d = 1; d < 16; d <<= 1) sm += __shfl_xor(sm, d, 32);
      lrow[r] = lrow[r] * al + sm;
      mrow[r] = nm;
      p0v[r] = p0; p1v[r] = p1;
#pragma unroll
      for (int ct = 0; ct < 4; ++ct) o[ct][r] *= al;
    }

    // C-layout -> A-layout via LDS (same wave; DS is in-order, fence compiler)
#pragma unroll
    for (int r = 0; r < 8; ++r) {
      ptile[wv][r + sel * 8][n]      = (half_t)p0v[r];
      ptile[wv][r + sel * 8][16 + n] = (half_t)p1v[r];
    }
    asm volatile("s_wait_dscnt 0" ::: "memory");

    // preload all V fragments, then the 4 PV WMMAs
    v16h pa = frag_a_f16(&ptile[wv][0][0], 32, 0, lane);
    v16h vb0 = frag_b_t(Vt + (long long)0  * SEQL, SEQL, kbase, lane);
    v16h vb1 = frag_b_t(Vt + (long long)16 * SEQL, SEQL, kbase, lane);
    v16h vb2 = frag_b_t(Vt + (long long)32 * SEQL, SEQL, kbase, lane);
    v16h vb3 = frag_b_t(Vt + (long long)48 * SEQL, SEQL, kbase, lane);
    o[0] = wmma16(pa, vb0, o[0]);
    o[1] = wmma16(pa, vb1, o[1]);
    o[2] = wmma16(pa, vb2, o[2]);
    o[3] = wmma16(pa, vb3, o[3]);
  }

  // epilogue: divide by row sums, write multi-head [b][s][h*64+dv] as f16
#pragma unroll
  for (int ct = 0; ct < 4; ++ct) {
#pragma unroll
    for (int r = 0; r < 8; ++r) {
      int row = q0 + r + sel * 8;
      float v = o[ct][r] / lrow[r];
      multi[((long long)b * SEQL + row) * DM + h * DK + ct * 16 + n] = (half_t)v;
    }
  }
}

// ---------- Kernel 4: output projection (async->LDS weights) -> fp32 ----------
__global__ __launch_bounds__(256) void k_oproj(const half_t* __restrict__ multi,
                                               const half_t* __restrict__ woT,
                                               float* __restrict__ out) {
  __shared__ half_t wlds[2][DK][64];           // 16 KB double buffer
  int tid  = threadIdx.x;
  int lane = tid & 31;
  int wv   = tid >> 5;
  int gw = blockIdx.x * 8 + wv;
  int rt = gw % (NROW / 16);
  int cg = gw / (NROW / 16);                   // 12 groups of 4 col-tiles (uniform per block)

  const half_t* wT = woT + (long long)cg * DK * DM;   // 64 output cols for this group
  const half_t* arow = multi + (long long)rt * 16 * DM;

  int cn = tid >> 2;
  int cp = (tid & 3) * 16;

  v8f c[4];
#pragma unroll
  for (int ct = 0; ct < 4; ++ct)
#pragma unroll
    for (int r = 0; r < 8; ++r) c[ct][r] = 0.f;

  async_ld_b128(&wlds[0][cn][cp],     wT + (long long)cn * DM + cp);
  async_ld_b128(&wlds[0][cn][cp + 8], wT + (long long)cn * DM + cp + 8);

  for (int s = 0; s < NSTAGE; ++s) {
    int buf = s & 1;
    wait_async0();
    __syncthreads();
    if (s + 1 < NSTAGE) {
      const half_t* g = wT + (long long)cn * DM + (s + 1) * 64 + cp;
      async_ld_b128(&wlds[buf ^ 1][cn][cp],     g);
      async_ld_b128(&wlds[buf ^ 1][cn][cp + 8], g + 8);
    }
#pragma unroll
    for (int kk = 0; kk < 2; ++kk) {
      v16h a = frag_a_f16(arow, DM, s * 64 + kk * 32, lane);
      v16h b0 = frag_b_lds(&wlds[buf][0][0],  kk * 32, lane);
      v16h b1 = frag_b_lds(&wlds[buf][16][0], kk * 32, lane);
      v16h b2 = frag_b_lds(&wlds[buf][32][0], kk * 32, lane);
      v16h b3 = frag_b_lds(&wlds[buf][48][0], kk * 32, lane);
      c[0] = wmma16(a, b0, c[0]);
      c[1] = wmma16(a, b1, c[1]);
      c[2] = wmma16(a, b2, c[2]);
      c[3] = wmma16(a, b3, c[3]);
      SCHED_DS_THEN_WMMA();
    }
  }

  int n = lane & 15, sel = (lane >> 4) & 1;
#pragma unroll
  for (int ct = 0; ct < 4; ++ct)
#pragma unroll
    for (int r = 0; r < 8; ++r)
      out[((long long)rt * 16 + r + sel * 8) * DM + (cg * 4 + ct) * 16 + n] = c[ct][r];
}

// ---------- host launcher ----------
extern "C" void kernel_launch(void* const* d_in, const int* in_sizes, int n_in,
                              void* d_out, int out_size, void* d_ws, size_t ws_size,
                              hipStream_t stream) {
  const float* residual = (const float*)d_in[0];
  const float* W_Q = (const float*)d_in[1];
  const float* W_K = (const float*)d_in[2];
  const float* W_V = (const float*)d_in[3];
  const float* W_O = (const float*)d_in[4];
  float* out = (float*)d_out;

  // workspace layout (in halves)
  const long long WSZ = (long long)HH * DM * DK;   // 589824
  const long long QSZ = (long long)HH * NROW * DK; // 6291456
  half_t* ws  = (half_t*)d_ws;
  half_t* wqT = ws;
  half_t* wkT = wqT + WSZ;
  half_t* wvT = wkT + WSZ;
  half_t* woT = wvT + WSZ;                         // 768x768
  half_t* Qb  = woT + (long long)DM * DM;
  half_t* Kb  = Qb + QSZ;
  half_t* VTb = Kb + QSZ;
  half_t* mh  = VTb + QSZ;                         // multi-head f16 [8192][768]

  // 1) weight convert/transpose
  {
    long long tot = WSZ;
    int blocks = (int)((tot + 255) / 256);
    k_transpose<<<blocks, 256, 0, stream>>>(W_Q, wqT, DM, DK, tot);
    k_transpose<<<blocks, 256, 0, stream>>>(W_K, wkT, DM, DK, tot);
    k_transpose<<<blocks, 256, 0, stream>>>(W_V, wvT, DM, DK, tot);
    long long tot2 = (long long)DM * DM;
    k_transpose<<<(int)((tot2 + 255) / 256), 256, 0, stream>>>(W_O, woT, DM, DM, tot2);
  }
  // 2) QKV projections: 3*12*512 waves / 8 per block
  k_qkv<<<(3 * HH * (NROW / 16)) / 8, 256, 0, stream>>>(residual, wqT, wkT, wvT, Qb, Kb, VTb);
  // 3) flash attention: 12*4*128 waves / 8 per block
  k_flash<<<(HH * BB * (SEQL / 16)) / 8, 256, 0, stream>>>(Qb, Kb, VTb, mh);
  // 4) output projection: 512*12 waves / 8 per block
  k_oproj<<<((NROW / 16) * 12) / 8, 256, 0, stream>>>(mh, woT, out);
}
